// graph_transformer_15539191677674
// MI455X (gfx1250) — compile-verified, matrix operand order checked by reading, not run on hardware
//
#include <hip/hip_runtime.h>
#include <stdint.h>

typedef __bf16 bf16x16 __attribute__((ext_vector_type(16)));
typedef float  f32x8   __attribute__((ext_vector_type(8)));
typedef float  f32x4   __attribute__((ext_vector_type(4)));

#define D_MODEL 512
#define N_HEAD  8
#define HEAD_DIM 64
#define FFN 2048
#define N_LAYERS 2
#define LN_EPS 1e-5f

// ---------------- helpers ----------------

__device__ __forceinline__ unsigned short f2bf(float f) {
  unsigned u = __float_as_uint(f);
  u = u + 0x7fffu + ((u >> 16) & 1u);          // round to nearest even
  return (unsigned short)(u >> 16);
}

// Fragment-ordered layout for 16-bit WMMA A operands (B uses same formula with
// n in place of row).  Element (row, col) of a [M x K] matrix, KT = K/32:
//   tile mt = row/16, ktile kt = col/32
//   lane    = (row%16) + 16*hi         with hi = (col%32 / 8) & 1
//   idx i   = 8*((col%32)/16) + (col%8)
// buffer index = ((mt*KT + kt)*32 + lane)*16 + i
__device__ __forceinline__ size_t fragIdx(int row, int col, int KT) {
  int mt = row >> 4, rm = row & 15;
  int kt = col >> 5, kk = col & 31;
  int hi = (kk >> 3) & 1;
  int lane = rm | (hi << 4);
  int i = ((kk >> 4) << 3) | (kk & 7);
  return (((size_t)(mt * KT + kt) * 32 + lane) * 16 + i);
}

// monotonic float<->uint mapping so unsigned atomicMax == float max
__device__ __forceinline__ unsigned f2mono(float f) {
  unsigned u = __float_as_uint(f);
  return (u & 0x80000000u) ? ~u : (u | 0x80000000u);
}
__device__ __forceinline__ float mono2f(unsigned t) {
  unsigned u = (t & 0x80000000u) ? (t & 0x7fffffffu) : ~t;
  return __uint_as_float(u);
}

// ---------------- swizzle kernels ----------------

// A operand: X f32 [M,K] row-major -> bf16 fragments
__global__ void k_swizzleA(const float* __restrict__ X,
                           unsigned short* __restrict__ F, int M, int K) {
  int idx = blockIdx.x * blockDim.x + threadIdx.x;
  int total = M * K;
  if (idx >= total) return;
  int row = idx / K, col = idx - row * K;
  F[fragIdx(row, col, K >> 5)] = f2bf(X[idx]);
}

// B operand: weight W f32 [Dout, Din] row-major; B(k,n) = W[n,k].
__global__ void k_swizzleB(const float* __restrict__ W,
                           unsigned short* __restrict__ F, int Dout, int Din) {
  int idx = blockIdx.x * blockDim.x + threadIdx.x;
  int total = Dout * Din;
  if (idx >= total) return;
  int n = idx / Din, k = idx - n * Din;
  F[fragIdx(n, k, Din >> 5)] = f2bf(W[idx]);   // same formula, n plays "row"
}

// ---------------- WMMA GEMM ----------------
// out[M,Nout] = (Afrag[M,K] @ Bfrag[K,Nout] + bias) * scale ; optional relu.
// Optionally writes f32 output and/or bf16 swizzled fragments (KT2 = Nout/32).
// Block: 256 threads = 8 waves (2x4). Wave tile: 32x64 (2x4 16x16 subtiles).
// KT = K/32 is a template constant so all six fragment addresses in the hot
// loop are base + immediate-offset (strides KT*1024/2048/3072 bytes), with a
// single pointer bump per 8 unrolled K-steps.
template <int KT>
__global__ void __launch_bounds__(256) k_gemm(
    const unsigned short* __restrict__ Af,
    const unsigned short* __restrict__ Bf,
    const float* __restrict__ bias,
    float* __restrict__ outF32,
    unsigned short* __restrict__ outFrag,
    int M, int Nout, float scale, int relu, int KT2) {
  const int wid  = threadIdx.x >> 5;
  const int lid  = threadIdx.x & 31;
  const int mBase = blockIdx.x * 64 + (wid >> 2) * 32;
  const int nBase = blockIdx.y * 256 + (wid & 3) * 64;
  if (mBase >= M || nBase >= Nout) return;

  const int mt0 = mBase >> 4, nt0 = nBase >> 4;
  // byte pointers: fragment for (tile, kt, lane) is 32 bytes
  const char* pa = (const char*)Af + ((size_t)mt0 * KT * 32 + lid) * 32;
  const char* pb = (const char*)Bf + ((size_t)nt0 * KT * 32 + lid) * 32;
  constexpr size_t TS = (size_t)KT * 1024;   // bytes per 16-row tile of frags

  f32x8 acc[2][4];
  const f32x8 zero = {0.f, 0.f, 0.f, 0.f, 0.f, 0.f, 0.f, 0.f};
#pragma unroll
  for (int i = 0; i < 2; ++i)
#pragma unroll
    for (int j = 0; j < 4; ++j) acc[i][j] = zero;

#pragma unroll 8
  for (int kt = 0; kt < KT; ++kt) {
    bf16x16 a0 = *(const bf16x16*)(pa);
    bf16x16 a1 = *(const bf16x16*)(pa + TS);
    bf16x16 b0 = *(const bf16x16*)(pb);
    bf16x16 b1 = *(const bf16x16*)(pb + TS);
    bf16x16 b2 = *(const bf16x16*)(pb + 2 * TS);
    bf16x16 b3 = *(const bf16x16*)(pb + 3 * TS);
    acc[0][0] = __builtin_amdgcn_wmma_f32_16x16x32_bf16(false, a0, false, b0, (short)0, acc[0][0], false, false);
    acc[1][0] = __builtin_amdgcn_wmma_f32_16x16x32_bf16(false, a1, false, b0, (short)0, acc[1][0], false, false);
    acc[0][1] = __builtin_amdgcn_wmma_f32_16x16x32_bf16(false, a0, false, b1, (short)0, acc[0][1], false, false);
    acc[1][1] = __builtin_amdgcn_wmma_f32_16x16x32_bf16(false, a1, false, b1, (short)0, acc[1][1], false, false);
    acc[0][2] = __builtin_amdgcn_wmma_f32_16x16x32_bf16(false, a0, false, b2, (short)0, acc[0][2], false, false);
    acc[1][2] = __builtin_amdgcn_wmma_f32_16x16x32_bf16(false, a1, false, b2, (short)0, acc[1][2], false, false);
    acc[0][3] = __builtin_amdgcn_wmma_f32_16x16x32_bf16(false, a0, false, b3, (short)0, acc[0][3], false, false);
    acc[1][3] = __builtin_amdgcn_wmma_f32_16x16x32_bf16(false, a1, false, b3, (short)0, acc[1][3], false, false);
    pa += 1024;   // one kt step = 32 lanes * 32 bytes
    pb += 1024;
  }

  // Epilogue.  C/D layout: lane n = lid&15, vgpr r -> m = r + 8*(lid>>4)
  const int n15 = lid & 15, hi = lid >> 4;
#pragma unroll
  for (int mi = 0; mi < 2; ++mi) {
#pragma unroll
    for (int j = 0; j < 4; ++j) {
      const int ncol = nBase + j * 16 + n15;
      const float bv = bias ? bias[ncol] : 0.f;
#pragma unroll
      for (int r = 0; r < 8; ++r) {
        const int mrow = mBase + mi * 16 + r + hi * 8;
        if (mrow >= M) continue;
        float v = (acc[mi][j][r] + bv) * scale;
        if (relu) v = fmaxf(v, 0.f);
        if (outF32)  outF32[(size_t)mrow * Nout + ncol] = v;
        if (outFrag) outFrag[fragIdx(mrow, ncol, KT2)] = f2bf(v);
      }
    }
  }
}

// ---------------- attention kernels ----------------

// one wave per edge: scores[e,h] = dot64(q[tgt[e],h,:], k[src[e],h,:])
__global__ void k_scores(const float* __restrict__ q, const float* __restrict__ k,
                         const int* __restrict__ srcI, const int* __restrict__ tgtI,
                         float* __restrict__ scores, int E) {
  int e = (blockIdx.x * blockDim.x + threadIdx.x) >> 5;
  int lid = threadIdx.x & 31;
  if (e >= E) return;
  const int t = tgtI[e], s = srcI[e];
  const f32x4* qp = (const f32x4*)(q + (size_t)t * D_MODEL) + lid * 4;
  const f32x4* kp = (const f32x4*)(k + (size_t)s * D_MODEL) + lid * 4;
  float acc = 0.f;
#pragma unroll
  for (int i = 0; i < 4; ++i) {
    f32x4 a = qp[i], b = kp[i];
    acc += a[0] * b[0] + a[1] * b[1] + a[2] * b[2] + a[3] * b[3];
  }
  acc += __shfl_xor(acc, 1, 32);   // lane group of 4 == one head (64 elems)
  acc += __shfl_xor(acc, 2, 32);
  if ((lid & 3) == 0) scores[(size_t)e * N_HEAD + (lid >> 2)] = acc;
}

__global__ void k_segmax(const float* __restrict__ scores,
                         const int* __restrict__ tgtI,
                         unsigned* __restrict__ m, int EH) {
  int idx = blockIdx.x * blockDim.x + threadIdx.x;
  if (idx >= EH) return;
  int e = idx >> 3, h = idx & 7;
  atomicMax(&m[(size_t)tgtI[e] * N_HEAD + h], f2mono(scores[idx]));
}

__global__ void k_softmax_p(float* __restrict__ scores, const int* __restrict__ tgtI,
                            const unsigned* __restrict__ m, float* __restrict__ s,
                            int EH) {
  int idx = blockIdx.x * blockDim.x + threadIdx.x;
  if (idx >= EH) return;
  int e = idx >> 3, h = idx & 7;
  size_t seg = (size_t)tgtI[e] * N_HEAD + h;
  float p = expf(scores[idx] - mono2f(m[seg]));
  scores[idx] = p;                 // reuse buffer for p
  atomicAdd(&s[seg], p);
}

__global__ void k_agg(const float* __restrict__ p, const float* __restrict__ s,
                      const float* __restrict__ v, const int* __restrict__ srcI,
                      const int* __restrict__ tgtI, float* __restrict__ agg, int EH) {
  int idx = blockIdx.x * blockDim.x + threadIdx.x;
  if (idx >= EH) return;
  int e = idx >> 3, h = idx & 7;
  const int t = tgtI[e], sr = srcI[e];
  const float w = p[idx] / (s[(size_t)t * N_HEAD + h] + 1e-16f);
  const float* vr = v + (size_t)sr * D_MODEL + h * HEAD_DIM;
  float* ar = agg + (size_t)t * D_MODEL + h * HEAD_DIM;
#pragma unroll
  for (int i = 0; i < HEAD_DIM; i += 4) {
    f32x4 vv = *(const f32x4*)(vr + i);
    atomicAdd(ar + i + 0, w * vv[0]);
    atomicAdd(ar + i + 1, w * vv[1]);
    atomicAdd(ar + i + 2, w * vv[2]);
    atomicAdd(ar + i + 3, w * vv[3]);
  }
}

// ---------------- batchnorm (training-mode batch stats) ----------------

#define BN_RPB 100
__global__ void k_bnstats(const float* __restrict__ x, const float* __restrict__ y,
                          float* __restrict__ sum, float* __restrict__ sumsq, int Nrows) {
  const int c = threadIdx.x;                 // 256 threads -> cols c, c+256
  const int r0 = blockIdx.x * BN_RPB;
  const int r1 = min(r0 + BN_RPB, Nrows);
  float s0 = 0.f, s1 = 0.f, q0 = 0.f, q1 = 0.f;
  for (int r = r0; r < r1; ++r) {
    size_t base = (size_t)r * D_MODEL;
    float a = x[base + c] + y[base + c];
    float b = x[base + c + 256] + y[base + c + 256];
    s0 += a; q0 += a * a;
    s1 += b; q1 += b * b;
  }
  atomicAdd(&sum[c], s0);       atomicAdd(&sumsq[c], q0);
  atomicAdd(&sum[c + 256], s1); atomicAdd(&sumsq[c + 256], q1);
}

__global__ void k_bnfin(float* __restrict__ sum, float* __restrict__ sumsq, int n) {
  int c = threadIdx.x;
  if (c >= D_MODEL) return;
  float mu = sum[c] / (float)n;
  float var = sumsq[c] / (float)n - mu * mu;
  sum[c] = mu;
  sumsq[c] = rsqrtf(var + LN_EPS);
}

// x = BN(x + y); also emit bf16 fragments for the next GEMM (K = 512, KT = 16)
__global__ void k_bnapply(float* __restrict__ x, const float* __restrict__ y,
                          const float* __restrict__ mu, const float* __restrict__ rs,
                          const float* __restrict__ g, const float* __restrict__ b,
                          unsigned short* __restrict__ frag, int Nrows) {
  int idx = blockIdx.x * blockDim.x + threadIdx.x;
  if (idx >= Nrows * D_MODEL) return;
  int c = idx & (D_MODEL - 1), r = idx >> 9;
  float v = (x[idx] + y[idx] - mu[c]) * rs[c] * g[c] + b[c];
  x[idx] = v;
  frag[fragIdx(r, c, D_MODEL >> 5)] = f2bf(v);
}

// ---------------- final layernorm (wave per row) ----------------

__global__ void k_layernorm(const float* __restrict__ x, const float* __restrict__ g,
                            const float* __restrict__ b, float* __restrict__ out,
                            int Nrows) {
  int row = (blockIdx.x * blockDim.x + threadIdx.x) >> 5;
  int lid = threadIdx.x & 31;
  if (row >= Nrows) return;
  const f32x4* rp = (const f32x4*)(x + (size_t)row * D_MODEL) + lid * 4;
  f32x4 vv[4];
  float s = 0.f;
#pragma unroll
  for (int i = 0; i < 4; ++i) {
    vv[i] = rp[i];
    s += vv[i][0] + vv[i][1] + vv[i][2] + vv[i][3];
  }
#pragma unroll
  for (int o = 16; o > 0; o >>= 1) s += __shfl_xor(s, o, 32);
  const float mu = s * (1.f / D_MODEL);
  float q = 0.f;
#pragma unroll
  for (int i = 0; i < 4; ++i)
#pragma unroll
    for (int j = 0; j < 4; ++j) { float d = vv[i][j] - mu; q += d * d; }
#pragma unroll
  for (int o = 16; o > 0; o >>= 1) q += __shfl_xor(q, o, 32);
  const float rsig = rsqrtf(q * (1.f / D_MODEL) + LN_EPS);
  const int base = lid * 16;
#pragma unroll
  for (int i = 0; i < 4; ++i)
#pragma unroll
    for (int j = 0; j < 4; ++j) {
      int c = base + i * 4 + j;
      out[(size_t)row * D_MODEL + c] = (vv[i][j] - mu) * rsig * g[c] + b[c];
    }
}

// ---------------- host orchestration ----------------

static inline void gemm_launch(const unsigned short* Af, const unsigned short* Bf,
                               const float* bias, float* o32, unsigned short* ofrag,
                               int M, int K, int Nout, float scale, int relu,
                               int KT2, hipStream_t st) {
  dim3 grid((M + 63) / 64, Nout / 256);
  if (K == 512) {
    k_gemm<16><<<grid, 256, 0, st>>>(Af, Bf, bias, o32, ofrag, M, Nout, scale, relu, KT2);
  } else { // K == 2048
    k_gemm<64><<<grid, 256, 0, st>>>(Af, Bf, bias, o32, ofrag, M, Nout, scale, relu, KT2);
  }
}

extern "C" void kernel_launch(void* const* d_in, const int* in_sizes, int n_in,
                              void* d_out, int out_size, void* d_ws, size_t ws_size,
                              hipStream_t stream) {
  const float* src   = (const float*)d_in[0];
  const int*   eidx  = (const int*)d_in[1];
  const float* Wq    = (const float*)d_in[2];
  const float* bq    = (const float*)d_in[3];
  const float* Wk    = (const float*)d_in[4];
  const float* Wv    = (const float*)d_in[5];
  const float* Wo    = (const float*)d_in[6];
  const float* bo    = (const float*)d_in[7];
  const float* W1    = (const float*)d_in[8];
  const float* b1    = (const float*)d_in[9];
  const float* W2    = (const float*)d_in[10];
  const float* b2    = (const float*)d_in[11];
  const float* g1    = (const float*)d_in[12];
  const float* beta1 = (const float*)d_in[13];
  const float* g2    = (const float*)d_in[14];
  const float* beta2 = (const float*)d_in[15];
  const float* ln_g  = (const float*)d_in[16];
  const float* ln_b  = (const float*)d_in[17];

  const int N  = in_sizes[0] / D_MODEL;
  const int E  = in_sizes[1] / 2;
  const int EH = E * N_HEAD;
  const int* srcI = eidx;        // edge_index[0]
  const int* tgtI = eidx + E;    // edge_index[1]

  // ---- workspace carve-out (256B aligned) ----
  size_t off = 0;
  auto carve = [&](size_t bytes) -> char* {
    char* p = (char*)d_ws + off;
    off = (off + bytes + 255) & ~(size_t)255;
    return p;
  };
  const size_t xb = (size_t)N * D_MODEL * sizeof(float);
  float*          xbuf  = (float*)carve(xb);
  float*          qbuf  = (float*)carve(xb);          // q, later attn-out
  float*          kbuf  = (float*)carve(xb);          // k, later ffn-out
  float*          vbuf  = (float*)carve(xb);
  float*          agg   = (float*)carve(xb);
  float*          scor  = (float*)carve((size_t)EH * sizeof(float)); // scores -> p
  unsigned*       mbuf  = (unsigned*)carve((size_t)N * N_HEAD * sizeof(unsigned));
  float*          sbuf  = (float*)carve((size_t)N * N_HEAD * sizeof(float));
  unsigned short* xfrag = (unsigned short*)carve((size_t)N * D_MODEL * 2);
  const int CH = 2048;                                 // FFN row chunk (mult of 32)
  unsigned short* midfrag = (unsigned short*)carve((size_t)CH * FFN * 2);
  unsigned short* wfrag[N_LAYERS][6];
  for (int l = 0; l < N_LAYERS; ++l) {
    for (int w = 0; w < 4; ++w)
      wfrag[l][w] = (unsigned short*)carve((size_t)D_MODEL * D_MODEL * 2);
    wfrag[l][4] = (unsigned short*)carve((size_t)FFN * D_MODEL * 2); // W1
    wfrag[l][5] = (unsigned short*)carve((size_t)D_MODEL * FFN * 2); // W2
  }
  float* bnsum   = (float*)carve(D_MODEL * sizeof(float));
  float* bnsumsq = (float*)carve(D_MODEL * sizeof(float));
  (void)ws_size; (void)n_in; (void)out_size;

  const int TPB = 256;
  auto blocks = [](long long n, int t) { return (unsigned)((n + t - 1) / t); };

  // ---- one-time per call: weight swizzles (deterministic, no caching) ----
  for (int l = 0; l < N_LAYERS; ++l) {
    const size_t wdd = (size_t)l * D_MODEL * D_MODEL;
    const size_t wfd = (size_t)l * FFN * D_MODEL;
    k_swizzleB<<<blocks((long long)D_MODEL * D_MODEL, TPB), TPB, 0, stream>>>(Wq + wdd, wfrag[l][0], D_MODEL, D_MODEL);
    k_swizzleB<<<blocks((long long)D_MODEL * D_MODEL, TPB), TPB, 0, stream>>>(Wk + wdd, wfrag[l][1], D_MODEL, D_MODEL);
    k_swizzleB<<<blocks((long long)D_MODEL * D_MODEL, TPB), TPB, 0, stream>>>(Wv + wdd, wfrag[l][2], D_MODEL, D_MODEL);
    k_swizzleB<<<blocks((long long)D_MODEL * D_MODEL, TPB), TPB, 0, stream>>>(Wo + wdd, wfrag[l][3], D_MODEL, D_MODEL);
    k_swizzleB<<<blocks((long long)FFN * D_MODEL, TPB), TPB, 0, stream>>>(W1 + wfd, wfrag[l][4], FFN, D_MODEL);
    k_swizzleB<<<blocks((long long)D_MODEL * FFN, TPB), TPB, 0, stream>>>(W2 + wfd, wfrag[l][5], D_MODEL, FFN);
  }

  // x = src (f32 copy for residuals) + bf16 fragments for first QKV GEMMs
  hipMemcpyAsync(xbuf, src, xb, hipMemcpyDeviceToDevice, stream);
  k_swizzleA<<<blocks((long long)N * D_MODEL, TPB), TPB, 0, stream>>>(src, xfrag, N, D_MODEL);

  const float qscale = 0.125f;   // 64^-0.5

  for (int l = 0; l < N_LAYERS; ++l) {
    const float* bq_l = bq + (size_t)l * D_MODEL;
    const float* bo_l = bo + (size_t)l * D_MODEL;
    const float* b1_l = b1 + (size_t)l * FFN;
    const float* b2_l = b2 + (size_t)l * D_MODEL;

    // ---- QKV projections (WMMA) ----
    gemm_launch(xfrag, wfrag[l][0], bq_l, qbuf, nullptr, N, D_MODEL, D_MODEL, qscale, 0, 0, stream);
    gemm_launch(xfrag, wfrag[l][1], nullptr, kbuf, nullptr, N, D_MODEL, D_MODEL, 1.f, 0, 0, stream);
    gemm_launch(xfrag, wfrag[l][2], nullptr, vbuf, nullptr, N, D_MODEL, D_MODEL, 1.f, 0, 0, stream);

    // ---- per-edge attention ----
    k_scores<<<blocks((long long)E * 32, TPB), TPB, 0, stream>>>(qbuf, kbuf, srcI, tgtI, scor, E);
    hipMemsetAsync(mbuf, 0, (size_t)N * N_HEAD * sizeof(unsigned), stream);
    hipMemsetAsync(sbuf, 0, (size_t)N * N_HEAD * sizeof(float), stream);
    k_segmax<<<blocks(EH, TPB), TPB, 0, stream>>>(scor, tgtI, mbuf, EH);
    k_softmax_p<<<blocks(EH, TPB), TPB, 0, stream>>>(scor, tgtI, mbuf, sbuf, EH);
    hipMemsetAsync(agg, 0, xb, stream);
    k_agg<<<blocks(EH, TPB), TPB, 0, stream>>>(scor, sbuf, vbuf, srcI, tgtI, agg, EH);

    // ---- output projection: attn = agg @ Wo^T + bo  (reuse qbuf) ----
    k_swizzleA<<<blocks((long long)N * D_MODEL, TPB), TPB, 0, stream>>>(agg, xfrag, N, D_MODEL);
    gemm_launch(xfrag, wfrag[l][3], bo_l, qbuf, nullptr, N, D_MODEL, D_MODEL, 1.f, 0, 0, stream);

    // ---- BN1(x + attn) -> xbuf (+ fragments for FF1) ----
    hipMemsetAsync(bnsum, 0, D_MODEL * sizeof(float), stream);
    hipMemsetAsync(bnsumsq, 0, D_MODEL * sizeof(float), stream);
    k_bnstats<<<blocks(N, BN_RPB), 256, 0, stream>>>(xbuf, qbuf, bnsum, bnsumsq, N);
    k_bnfin<<<1, D_MODEL, 0, stream>>>(bnsum, bnsumsq, N);
    k_bnapply<<<blocks((long long)N * D_MODEL, TPB), TPB, 0, stream>>>(xbuf, qbuf, bnsum, bnsumsq,
        g1 + (size_t)l * D_MODEL, beta1 + (size_t)l * D_MODEL, xfrag, N);

    // ---- FFN in row chunks; FF1 emits bf16 fragments directly ----
    for (int o = 0; o < N; o += CH) {
      const int M = min(CH, N - o);
      gemm_launch(xfrag + (size_t)o * D_MODEL, wfrag[l][4], b1_l,
                  nullptr, midfrag, M, D_MODEL, FFN, 1.f, 1, FFN >> 5, stream);
      gemm_launch(midfrag, wfrag[l][5], b2_l,
                  kbuf + (size_t)o * D_MODEL, nullptr, M, FFN, D_MODEL, 1.f, 0, 0, stream);
    }

    // ---- BN2(x + ff) -> xbuf (+ fragments for next layer's QKV) ----
    hipMemsetAsync(bnsum, 0, D_MODEL * sizeof(float), stream);
    hipMemsetAsync(bnsumsq, 0, D_MODEL * sizeof(float), stream);
    k_bnstats<<<blocks(N, BN_RPB), 256, 0, stream>>>(xbuf, kbuf, bnsum, bnsumsq, N);
    k_bnfin<<<1, D_MODEL, 0, stream>>>(bnsum, bnsumsq, N);
    k_bnapply<<<blocks((long long)N * D_MODEL, TPB), TPB, 0, stream>>>(xbuf, kbuf, bnsum, bnsumsq,
        g2 + (size_t)l * D_MODEL, beta2 + (size_t)l * D_MODEL, xfrag, N);
  }

  // ---- final LayerNorm -> d_out ----
  k_layernorm<<<blocks((long long)N * 32, TPB), TPB, 0, stream>>>(xbuf, ln_g, ln_b, (float*)d_out, N);
}
